// GCNLayer_34239479283727
// MI455X (gfx1250) — matrix operand order repaired and checked by reading, not compiled, not run on hardware
//
#include <hip/hip_runtime.h>

// GCN layer for MI455X (gfx1250), wave32 + WMMA bf16 hi/lo split + TDM staging.
//
// out = (hat_d ⊙ (A @ (hat_d ⊙ feature) + hat_d ⊙ feature)) @ W^T + b
//
// Roofline: A (1 GiB fp32) streams once from HBM = ~46 us floor @ 23.3 TB/s.
// Main GEMM (275 GFLOP) runs as 3x split-bf16 WMMA (hi*hi + lo*hi + hi*lo) to
// keep ~fp32 accuracy near the bf16-WMMA rate. dh (32 MB bf16 hi/lo) stays
// L2-resident; its per-step 2D tile is DMA'd into double-buffered LDS by the
// Tensor Data Mover (pad_enable gives the 80B padded row stride), overlapped
// with compute and fenced by s_wait_tensorcnt + one barrier per K-step.
// Workspace requirement: 64 MiB.

typedef __bf16 bf16_t;
typedef __attribute__((ext_vector_type(16))) __bf16 v16bf;
typedef __attribute__((ext_vector_type(8)))  __bf16 v8bf;
typedef __attribute__((ext_vector_type(8)))  float  v8f;
typedef __attribute__((ext_vector_type(4)))  float  v4f;
typedef __attribute__((ext_vector_type(4)))  unsigned int v4u;
typedef __attribute__((ext_vector_type(8)))  unsigned int v8u;

#define NN   16384
#define FIN  512
#define FOUT 256

__device__ __forceinline__ v8f wmma_bf16(v16bf a, v16bf b, v8f c) {
  return __builtin_amdgcn_wmma_f32_16x16x32_bf16(false, a, false, b, (short)0, c,
                                                 false, false);
}

__device__ __forceinline__ void split2(float x, bf16_t& hi, bf16_t& lo) {
  hi = (bf16_t)x;
  lo = (bf16_t)(x - (float)hi);
}

// ---------------------------------------------------------------------------
// Kernel 1: dhT[c][k] = split_bf16(hat_d[k] * feature[k][c])   (transposed)
// ---------------------------------------------------------------------------
__global__ __launch_bounds__(256)
void gcn_pack_dhT(const float* __restrict__ hat_d,
                  const float* __restrict__ feature,
                  bf16_t* __restrict__ dhT_hi,
                  bf16_t* __restrict__ dhT_lo)
{
  __shared__ bf16_t th[32][33];
  __shared__ bf16_t tl[32][33];
  const int r0 = blockIdx.x * 32;
  const int c0 = blockIdx.y * 32;
  const int t  = threadIdx.x;
  {
    const int ci = t & 31, rb = t >> 5;
    #pragma unroll
    for (int i = 0; i < 4; ++i) {
      const int r = rb + 8 * i;
      const float d = hat_d[r0 + r];
      const float v = d * feature[(size_t)(r0 + r) * FIN + (c0 + ci)];
      split2(v, th[r][ci], tl[r][ci]);
    }
  }
  __syncthreads();
  {
    const int ri = t & 31, cb = t >> 5;
    #pragma unroll
    for (int i = 0; i < 4; ++i) {
      const int c = cb + 8 * i;
      const size_t o = (size_t)(c0 + c) * NN + (r0 + ri);
      dhT_hi[o] = th[ri][c];
      dhT_lo[o] = tl[ri][c];
    }
  }
}

// ---------------------------------------------------------------------------
// TDM: load a 2D tile (32 k-elems x 512 lines, 2B elems, line stride NN elems)
// into LDS with +16B padding after each 64B line => 80B LDS row stride.
// D# per CDNA5 ISA ch.8: group0 = {count/type/addr}, group1 = {dims/strides}.
// ---------------------------------------------------------------------------
__device__ __forceinline__ void tdm_load_tile(unsigned lds_off, const void* gptr)
{
  const unsigned long long ga = (unsigned long long)(uintptr_t)gptr;
  v4u g0;
  g0[0] = 1u;                                        // count=1 (valid), user mode
  g0[1] = lds_off;                                   // lds_addr (bytes)
  g0[2] = (unsigned)ga;                              // global_addr[31:0]
  g0[3] = (unsigned)((ga >> 32) & 0x01FFFFFFull)     // global_addr[56:32]
        | (2u << 30);                                // type=2 ("image")
  v8u g1;
  g1[0] = (1u << 16)      // data_size = 1 -> 2 bytes
        | (1u << 20)      // pad_enable
        | (3u << 22)      // pad_interval: 3 -> every 16 DWORDs (64B)
        | (3u << 25);     // pad_amount:   3 -> 4 DWORDs (16B)
  g1[1] = (32u << 16);    // tensor_dim0 = 32 (bits 79:48, low half)
  g1[2] = (512u << 16);   // tensor_dim0 hi=0 | tensor_dim1 = 512 (low half)
  g1[3] = (32u << 16);    // tensor_dim1 hi=0 | tile_dim0 = 32
  g1[4] = 512u;           // tile_dim1 = 512 | tile_dim2 = 0 (2D)
  g1[5] = 16384u;         // tensor_dim0_stride[31:0] = NN elements
  g1[6] = 0u;             // stride hi | tensor_dim1_stride (unused, 2D)
  g1[7] = 0u;
  asm volatile("tensor_load_to_lds %0, %1" :: "s"(g0), "s"(g1) : "memory");
}

// ---------------------------------------------------------------------------
// Kernel 2: S = A @ dh ; h = hat_d ⊙ (S + hat_d ⊙ feature) -> bf16 hi/lo.
// Block: 64 rows x 512 cols, 8 waves (2x4 of 32x128). A fragments load fp32
// straight from global and split in-register; dhT tiles are TDM-DMA'd into
// double-buffered LDS (exactly 160 KiB).
// ---------------------------------------------------------------------------
#define G1_BR     64
#define G1_BK     32
#define G1_KSTEPS (NN / G1_BK)
#define BSTR      40   // padded LDS row stride in elements (80B)

__global__ __launch_bounds__(256)
void gcn_aggregate(const float* __restrict__ A,
                   const bf16_t* __restrict__ dhT_hi,
                   const bf16_t* __restrict__ dhT_lo,
                   const float* __restrict__ hat_d,
                   const float* __restrict__ feature,
                   bf16_t* __restrict__ h_hi,
                   bf16_t* __restrict__ h_lo)
{
  __shared__ __align__(16) bf16_t BhL[2][FIN * BSTR];  // 2 x 40 KiB
  __shared__ __align__(16) bf16_t BlL[2][FIN * BSTR];  // 2 x 40 KiB

  const int t    = threadIdx.x;
  const int lane = t & 31;
  const int wave = t >> 5;
  const int wr   = wave >> 2;   // 0..1  -> 32-row group
  const int wc   = wave & 3;    // 0..3  -> 128-col group
  const int lm   = lane & 15;
  const int lh   = lane >> 4;
  const int row0 = blockIdx.x * G1_BR;

  v8f acc[2][8] = {};

  // A prefetch registers: per m-fragment, two 8-float chunks (contiguous K:
  // K {kb..kb+7, kb+8..kb+15}, kb = lh*16).
  v8f aNxt[2][2];
  auto loadA = [&](int k0) {
    #pragma unroll
    for (int mi = 0; mi < 2; ++mi) {
      const float* p = A + (size_t)(row0 + wr * 32 + mi * 16 + lm) * NN
                         + k0 + lh * 16;
      aNxt[mi][0] = *(const v8f*)p;
      aNxt[mi][1] = *(const v8f*)(p + 8);
    }
  };

  const unsigned bh0 = (unsigned)(uintptr_t)&BhL[0][0];
  const unsigned bh1 = (unsigned)(uintptr_t)&BhL[1][0];
  const unsigned bl0 = (unsigned)(uintptr_t)&BlL[0][0];
  const unsigned bl1 = (unsigned)(uintptr_t)&BlL[1][0];

  // prologue: A regs for step 0 + TDM of B tile 0 into buffer 0
  loadA(0);
  if (wave == 0) {
    tdm_load_tile(bh0, dhT_hi);
    tdm_load_tile(bl0, dhT_lo);
    __builtin_amdgcn_s_wait_tensorcnt(0);
  }
  __syncthreads();

  for (int kk = 0; kk < G1_KSTEPS; ++kk) {
    const int cur = kk & 1;

    // convert this step's A (fp32 -> bf16 hi/lo fragments)
    v16bf ah[2], al[2];
    #pragma unroll
    for (int mi = 0; mi < 2; ++mi) {
      #pragma unroll
      for (int j = 0; j < 8; ++j) {
        bf16_t h0, l0, h1, l1;
        split2(aNxt[mi][0][j], h0, l0);
        split2(aNxt[mi][1][j], h1, l1);
        ah[mi][j] = h0; al[mi][j] = l0;
        ah[mi][8 + j] = h1; al[mi][8 + j] = l1;
      }
    }

    // prefetch next step: A into regs, B via TDM into the other buffer
    if (kk + 1 < G1_KSTEPS) {
      const int k0n = (kk + 1) * G1_BK;
      loadA(k0n);
      if (wave == 0) {
        tdm_load_tile(cur ? bh0 : bh1, dhT_hi + k0n);
        tdm_load_tile(cur ? bl0 : bl1, dhT_lo + k0n);
      }
    }

    // WMMA over the current B buffer
    const bf16_t* Bh = &BhL[cur][0];
    const bf16_t* Bl = &BlL[cur][0];
    #pragma unroll
    for (int ni = 0; ni < 8; ++ni) {
      const int n = wc * 128 + ni * 16 + lm;
      v16bf bh = *(const v16bf*)(Bh + n * BSTR + lh * 16);
      v16bf bl = *(const v16bf*)(Bl + n * BSTR + lh * 16);
      #pragma unroll
      for (int mi = 0; mi < 2; ++mi) {
        acc[mi][ni] = wmma_bf16(ah[mi], bh, acc[mi][ni]);
        acc[mi][ni] = wmma_bf16(al[mi], bh, acc[mi][ni]);
        acc[mi][ni] = wmma_bf16(ah[mi], bl, acc[mi][ni]);
      }
    }

    // publish the next buffer: DMA must be complete before anyone reads it
    if (wave == 0 && kk + 1 < G1_KSTEPS) __builtin_amdgcn_s_wait_tensorcnt(0);
    __syncthreads();
  }

  // epilogue: h = d*(S + d*feature), re-split to bf16 hi/lo for the linear layer
  #pragma unroll
  for (int mi = 0; mi < 2; ++mi) {
    #pragma unroll
    for (int ni = 0; ni < 8; ++ni) {
      const int col = wc * 128 + ni * 16 + lm;
      #pragma unroll
      for (int r = 0; r < 8; ++r) {
        const int row = row0 + wr * 32 + mi * 16 + lh * 8 + r;
        const float d = hat_d[row];
        const float hv = d * (acc[mi][ni][r] + d * feature[(size_t)row * FIN + col]);
        bf16_t hi, lo; split2(hv, hi, lo);
        const size_t o = (size_t)row * FIN + col;
        h_hi[o] = hi;
        h_lo[o] = lo;
      }
    }
  }
}

// ---------------------------------------------------------------------------
// Kernel 3: out = h @ W^T + b   (16384x512 @ 512x256, L2-resident, WMMA bf16
// split). h is [row][k] (native A layout); W is [n][k] (native B layout),
// converted to bf16 hi/lo in-register.
// ---------------------------------------------------------------------------
__global__ __launch_bounds__(256)
void gcn_linear(const bf16_t* __restrict__ h_hi,
                const bf16_t* __restrict__ h_lo,
                const float* __restrict__ W,
                const float* __restrict__ bias,
                float* __restrict__ out)
{
  const int t    = threadIdx.x;
  const int lane = t & 31;
  const int wave = t >> 5;
  const int wr   = wave >> 2;   // 0..1 -> 32-row group
  const int wc   = wave & 3;    // 0..3 -> 64-col group
  const int lm   = lane & 15;
  const int lh   = lane >> 4;
  const int row0 = blockIdx.x * 64;

  v8f acc[2][4] = {};

  for (int k0 = 0; k0 < FIN; k0 += 32) {
    v16bf ah[2], al[2];
    #pragma unroll
    for (int mi = 0; mi < 2; ++mi) {
      const size_t rb = (size_t)(row0 + wr * 32 + mi * 16 + lm) * FIN + k0 + lh * 8;
      v8bf a0 = *(const v8bf*)(h_hi + rb);
      v8bf a1 = *(const v8bf*)(h_hi + rb + 16);
      v8bf c0 = *(const v8bf*)(h_lo + rb);
      v8bf c1 = *(const v8bf*)(h_lo + rb + 16);
      ah[mi] = __builtin_shufflevector(a0, a1, 0,1,2,3,4,5,6,7,8,9,10,11,12,13,14,15);
      al[mi] = __builtin_shufflevector(c0, c1, 0,1,2,3,4,5,6,7,8,9,10,11,12,13,14,15);
    }
    #pragma unroll
    for (int ni = 0; ni < 4; ++ni) {
      const int n = wc * 64 + ni * 16 + lm;
      const float* wp = W + (size_t)n * FIN + k0 + lh * 8;
      v16bf bh, bl;
      #pragma unroll
      for (int j = 0; j < 16; ++j) {
        bf16_t hi, lo; split2(wp[j + (j & 8)], hi, lo);
        bh[j] = hi; bl[j] = lo;
      }
      #pragma unroll
      for (int mi = 0; mi < 2; ++mi) {
        acc[mi][ni] = wmma_bf16(ah[mi], bh, acc[mi][ni]);
        acc[mi][ni] = wmma_bf16(al[mi], bh, acc[mi][ni]);
        acc[mi][ni] = wmma_bf16(ah[mi], bl, acc[mi][ni]);
      }
    }
  }

  #pragma unroll
  for (int mi = 0; mi < 2; ++mi) {
    #pragma unroll
    for (int ni = 0; ni < 4; ++ni) {
      const int col = wc * 64 + ni * 16 + lm;
      const float bb = bias[col];
      #pragma unroll
      for (int r = 0; r < 8; ++r) {
        const int row = row0 + wr * 32 + mi * 16 + lh * 8 + r;
        out[(size_t)row * FOUT + col] = acc[mi][ni][r] + bb;
      }
    }
  }
}

// ---------------------------------------------------------------------------
extern "C" void kernel_launch(void* const* d_in, const int* in_sizes, int n_in,
                              void* d_out, int out_size, void* d_ws, size_t ws_size,
                              hipStream_t stream)
{
  const float* A       = (const float*)d_in[0];
  const float* hat_d   = (const float*)d_in[1];
  const float* feature = (const float*)d_in[2];
  const float* W       = (const float*)d_in[3];
  const float* bias    = (const float*)d_in[4];
  float* out = (float*)d_out;

  // workspace layout: 4 x (16384*512) bf16 buffers = 64 MiB
  const size_t HALF = (size_t)NN * FIN;
  bf16_t* dhT_hi = (bf16_t*)d_ws;
  bf16_t* dhT_lo = dhT_hi + HALF;
  bf16_t* h_hi   = dhT_lo + HALF;
  bf16_t* h_lo   = h_hi + HALF;

  gcn_pack_dhT<<<dim3(NN / 32, FIN / 32), 256, 0, stream>>>(hat_d, feature,
                                                            dhT_hi, dhT_lo);
  gcn_aggregate<<<NN / G1_BR, 256, 0, stream>>>(A, dhT_hi, dhT_lo, hat_d,
                                                feature, h_hi, h_lo);
  gcn_linear<<<NN / 64, 256, 0, stream>>>(h_hi, h_lo, W, bias, out);

  (void)in_sizes; (void)n_in; (void)out_size; (void)ws_size;
}
